// STGNNRec_76982993813636
// MI455X (gfx1250) — matrix operand order repaired and checked by reading, not compiled, not run on hardware
//
#include <hip/hip_runtime.h>

#define UN      100000
#define IT_N    50000
#define DD      64
#define BB      8192
#define TT      30
#define E_UI    1600000
#define E_II    800000
#define LN_EPS  1e-5f
#define BEH_SC  0.35f
#define REG_CF  1e-4f

typedef __attribute__((ext_vector_type(16))) __bf16 v16bf;
typedef __attribute__((ext_vector_type(8)))  __bf16 v8bf;
typedef __attribute__((ext_vector_type(8)))  float  v8f;
typedef __attribute__((ext_vector_type(4)))  float  v4f;

// ---------------------------------------------------------------- WMMA core
static __device__ __forceinline__ v8f wmma_bf16(v16bf a, v16bf b, v8f c) {
  // 8 args: (neg_a, A, neg_b, B, c_mod, C, reuse_a, reuse_b)
  return __builtin_amdgcn_wmma_f32_16x16x32_bf16(false, a, false, b, (short)0, c,
                                                 false, false);
}

// Pre-swizzle a row-major f32 [64 x (CT*16)] weight matrix into LDS bf16
// B-fragments.  Fragment f = kt*CT + ct covers K rows kt*32..+31, cols ct*16..+15.
// Layout: frag[((f*32 + lane)*16) + i], h=lane>>4 (N=ct*16+lane%16):
//   i<8 -> K=kt*32+8h+i,  i>=8 -> K=kt*32+16+8h+(i-8).
static __device__ __forceinline__ void swizzle_B(const float* __restrict__ W, int ldw,
                                                 int CT, __bf16* frag, int tid, int nth) {
  int total = 2 * CT * 32 * 16;
  for (int x = tid; x < total; x += nth) {
    int i    = x & 15;
    int lane = (x >> 4) & 31;
    int f    = x >> 9;
    int kt   = f / CT;
    int ct   = f - kt * CT;
    int n    = ct * 16 + (lane & 15);
    int k    = kt * 32 + ((lane >> 4) << 3) + ((i >> 3) << 4) + (i & 7);
    frag[x]  = (__bf16)W[k * ldw + n];
  }
}

static __device__ __forceinline__ v16bf load_b_frag(const __bf16* frag, int f, int lane) {
  const v8bf* p = (const v8bf*)(frag + ((f * 32 + lane) << 4));
  v8bf lo = p[0], hi = p[1];
  return __builtin_shufflevector(lo, hi, 0, 1, 2, 3, 4, 5, 6, 7,
                                 8, 9, 10, 11, 12, 13, 14, 15);
}

// A-fragment (16 rows x 32 K) from row-major f32.  ISA 16-bit A layout:
// lanes 0-15: halves 0..7 = K k0+0..7,  halves 8..15 = K k0+16..23
// lanes 16-31: halves 0..7 = K k0+8..15, halves 8..15 = K k0+24..31
static __device__ __forceinline__ v16bf load_a_frag(const float* base, int ld, int k0,
                                                    int lane) {
  int row = lane & 15;
  int hi8 = (lane >> 4) << 3;
  const float* p = base + (size_t)row * ld + k0 + hi8;
  v4f x0 = *(const v4f*)(p);
  v4f x1 = *(const v4f*)(p + 4);
  v4f x2 = *(const v4f*)(p + 16);
  v4f x3 = *(const v4f*)(p + 20);
  v16bf a;
#pragma unroll
  for (int i = 0; i < 4; i++) {
    a[i]      = (__bf16)x0[i];
    a[4 + i]  = (__bf16)x1[i];
    a[8 + i]  = (__bf16)x2[i];
    a[12 + i] = (__bf16)x3[i];
  }
  return a;
}

static __device__ __forceinline__ v16bf load_a_frag_sum(const float* ba, const float* bb,
                                                        int ld, int k0, int lane) {
  int row = lane & 15;
  int hi8 = (lane >> 4) << 3;
  const float* p = ba + (size_t)row * ld + k0 + hi8;
  const float* q = bb + (size_t)row * ld + k0 + hi8;
  v4f x0 = *(const v4f*)(p)      + *(const v4f*)(q);
  v4f x1 = *(const v4f*)(p + 4)  + *(const v4f*)(q + 4);
  v4f x2 = *(const v4f*)(p + 16) + *(const v4f*)(q + 16);
  v4f x3 = *(const v4f*)(p + 20) + *(const v4f*)(q + 20);
  v16bf a;
#pragma unroll
  for (int i = 0; i < 4; i++) {
    a[i]      = (__bf16)x0[i];
    a[4 + i]  = (__bf16)x1[i];
    a[8 + i]  = (__bf16)x2[i];
    a[12 + i] = (__bf16)x3[i];
  }
  return a;
}

// Guaranteed single-instruction fp32 atomic add (no-return form, STOREcnt path).
// Avoids any toolchain CAS-loop lowering in the bandwidth-critical edge phase.
static __device__ __forceinline__ void atomAddF(float* p, float v) {
  asm volatile("global_atomic_add_f32 %0, %1, off"
               :
               : "v"(p), "v"(v)
               : "memory");
}

// ---------------------------------------------------------------- edge kernels
// agg_u[r] += v*it[c];  agg_i[c] += v*u[r]   (one wave per edge, float2/lane)
__global__ void __launch_bounds__(256) k_edges_ui(const float* __restrict__ u,
    const float* __restrict__ it, const float* __restrict__ vals,
    const int* __restrict__ rows, const int* __restrict__ cols,
    float* agg_u, float* agg_i, int ne) {
  long t = (long)blockIdx.x * 256 + threadIdx.x;
  int e = (int)(t >> 5);
  if (e >= ne) return;
  int lane = (int)(t & 31);
  float v  = vals[e];
  size_t r = (size_t)rows[e] * DD;
  size_t c = (size_t)cols[e] * DD;
  int o = lane * 2;
  float i0 = it[c + o], i1 = it[c + o + 1];
  float u0 = u[r + o],  u1 = u[r + o + 1];
  atomAddF(&agg_u[r + o],     v * i0);
  atomAddF(&agg_u[r + o + 1], v * i1);
  atomAddF(&agg_i[c + o],     v * u0);
  atomAddF(&agg_i[c + o + 1], v * u1);
}

// agg_i[rows[e]] += v*it[cols[e]]
__global__ void __launch_bounds__(256) k_edges_ii(const float* __restrict__ it,
    const float* __restrict__ vals, const int* __restrict__ rows,
    const int* __restrict__ cols, float* agg_i, int ne) {
  long t = (long)blockIdx.x * 256 + threadIdx.x;
  int e = (int)(t >> 5);
  if (e >= ne) return;
  int lane = (int)(t & 31);
  float v  = vals[e];
  size_t r = (size_t)rows[e] * DD;
  size_t c = (size_t)cols[e] * DD;
  int o = lane * 2;
  atomAddF(&agg_i[r + o],     v * it[c + o]);
  atomAddF(&agg_i[r + o + 1], v * it[c + o + 1]);
}

// ---------------------------------------------------------------- GNN transform
// agg = relu((src + agg) @ W + b)   (in place over agg; 16 rows / wave, WMMA)
__global__ void __launch_bounds__(128) k_transform(const float* __restrict__ src,
    float* agg, const float* __restrict__ W, const float* __restrict__ bias, int nrows) {
  __shared__ __attribute__((aligned(16))) __bf16 wfrag[2 * 4 * 32 * 16];
  __shared__ float sb[DD];
  int tid = threadIdx.x;
  swizzle_B(W, DD, 4, wfrag, tid, 128);
  if (tid < DD) sb[tid] = bias[tid];
  __syncthreads();
  int lane = tid & 31, wave = tid >> 5;
  int row0 = (blockIdx.x * 4 + wave) << 4;
  if (row0 >= nrows) return;
  const float* s0 = src + (size_t)row0 * DD;
  float* ap = agg + (size_t)row0 * DD;
  v16bf a0 = load_a_frag_sum(s0, ap, DD, 0, lane);
  v16bf a1 = load_a_frag_sum(s0, ap, DD, 32, lane);
  int rb = (lane >> 4) << 3;
#pragma unroll
  for (int ct = 0; ct < 4; ct++) {
    int col = ct * 16 + (lane & 15);
    float bv = sb[col];
    v8f c;
#pragma unroll
    for (int v = 0; v < 8; v++) c[v] = bv;
    c = wmma_bf16(a0, load_b_frag(wfrag, ct, lane), c);
    c = wmma_bf16(a1, load_b_frag(wfrag, 4 + ct, lane), c);
#pragma unroll
    for (int v = 0; v < 8; v++)
      ap[(size_t)(rb + v) * DD + col] = fmaxf(c[v], 0.f);
  }
}

// ---------------------------------------------------------------- seq_x build
// seqx[p] = LN( it[item[p]] + 0.35*beh[bh[p]] + relu(log1p(dt)*w1+b1) @ W2 + b2 )
__global__ void __launch_bounds__(128) k_seqx(const float* __restrict__ itemb,
    const float* __restrict__ behemb, const int* __restrict__ s_items,
    const int* __restrict__ s_beh, const float* __restrict__ delta,
    const float* __restrict__ w1, const float* __restrict__ b1,
    const float* __restrict__ W2, const float* __restrict__ b2,
    const float* __restrict__ lng, const float* __restrict__ lnb,
    float* __restrict__ seqx) {
  __shared__ __attribute__((aligned(16))) __bf16 wfrag[2 * 4 * 32 * 16];
  __shared__ float sw1[DD], sb1[DD], sb2[DD], sg[DD], sbb[DD];
  int tid = threadIdx.x;
  swizzle_B(W2, DD, 4, wfrag, tid, 128);
  if (tid < DD) {
    sw1[tid] = w1[tid]; sb1[tid] = b1[tid]; sb2[tid] = b2[tid];
    sg[tid] = lng[tid]; sbb[tid] = lnb[tid];
  }
  __syncthreads();
  int lane = tid & 31, wave = tid >> 5;
  int pos0 = (blockIdx.x * 4 + wave) << 4;
  if (pos0 >= BB * TT) return;
  int row = lane & 15;
  int hi8 = (lane >> 4) << 3;
  float tv = log1pf(delta[pos0 + row]);
  v16bf a0, a1;
#pragma unroll
  for (int i = 0; i < 8; i++) {
    int kA = hi8 + i, kB = 16 + hi8 + i;
    a0[i]     = (__bf16)fmaxf(tv * sw1[kA] + sb1[kA], 0.f);
    a0[8 + i] = (__bf16)fmaxf(tv * sw1[kB] + sb1[kB], 0.f);
    a1[i]     = (__bf16)fmaxf(tv * sw1[32 + kA] + sb1[32 + kA], 0.f);
    a1[8 + i] = (__bf16)fmaxf(tv * sw1[32 + kB] + sb1[32 + kB], 0.f);
  }
  int rb = (lane >> 4) << 3;
  v8f ctile[4];
#pragma unroll
  for (int ct = 0; ct < 4; ct++) {
    int col = ct * 16 + row;
    v8f c;
#pragma unroll
    for (int v = 0; v < 8; v++) {
      int p = pos0 + rb + v;
      c[v] = itemb[(size_t)s_items[p] * DD + col]
           + BEH_SC * behemb[(size_t)s_beh[p] * DD + col] + sb2[col];
    }
    c = wmma_bf16(a0, load_b_frag(wfrag, ct, lane), c);
    c = wmma_bf16(a1, load_b_frag(wfrag, 4 + ct, lane), c);
    ctile[ct] = c;
  }
  // LayerNorm per row: row lives in one VGPR index across 16 lanes x 4 tiles
#pragma unroll
  for (int v = 0; v < 8; v++) {
    float s = 0.f, s2 = 0.f;
#pragma unroll
    for (int ct = 0; ct < 4; ct++) { float x = ctile[ct][v]; s += x; s2 += x * x; }
#pragma unroll
    for (int m = 1; m < 16; m <<= 1) {
      s  += __shfl_xor(s, m, 32);
      s2 += __shfl_xor(s2, m, 32);
    }
    float mu  = s * (1.f / DD);
    float var = s2 * (1.f / DD) - mu * mu;
    float inv = rsqrtf(var + LN_EPS);
    size_t orow = (size_t)(pos0 + rb + v) * DD;
#pragma unroll
    for (int ct = 0; ct < 4; ct++) {
      int col = ct * 16 + row;
      seqx[orow + col] = (ctile[ct][v] - mu) * inv * sg[col] + sbb[col];
    }
  }
}

// ---------------------------------------------------------------- GRU (persistent)
__global__ void __launch_bounds__(128) k_gru(const float* __restrict__ seqx,
    const float* __restrict__ Wih, const float* __restrict__ Whh,
    const float* __restrict__ bih, const float* __restrict__ bhh,
    const int* __restrict__ slen, float* __restrict__ srep) {
  __shared__ __attribute__((aligned(16))) __bf16 fih[2 * 12 * 32 * 16];  // 24 KB
  __shared__ __attribute__((aligned(16))) __bf16 fhh[2 * 12 * 32 * 16];  // 24 KB
  __shared__ __attribute__((aligned(16))) float stage[4][16 * DD];       // 16 KB
  int tid = threadIdx.x;
  swizzle_B(Wih, 3 * DD, 12, fih, tid, 128);
  swizzle_B(Whh, 3 * DD, 12, fhh, tid, 128);
  __syncthreads();
  int lane = tid & 31, wave = tid >> 5;
  int b0 = (blockIdx.x * 4 + wave) << 4;
  if (b0 >= BB) return;
  float* stg = stage[wave];
  int row = lane & 15;
  int rb = (lane >> 4) << 3;
  float bI[12], bH[12];
#pragma unroll
  for (int cb = 0; cb < 12; cb++) { bI[cb] = bih[cb * 16 + row]; bH[cb] = bhh[cb * 16 + row]; }
  int sl[8];
#pragma unroll
  for (int v = 0; v < 8; v++) sl[v] = slen[b0 + rb + v];
  v8f h[4];
#pragma unroll
  for (int ct = 0; ct < 4; ct++)
#pragma unroll
    for (int v = 0; v < 8; v++) h[ct][v] = 0.f;

  for (int t = 0; t < TT; t++) {
    const float* xb = seqx + ((size_t)b0 * TT + t) * DD;
    v16bf xa0 = load_a_frag(xb, TT * DD, 0, lane);
    v16bf xa1 = load_a_frag(xb, TT * DD, 32, lane);
    // re-layout h (C tiles) -> A fragments via per-wave LDS staging (wave-in-order DS)
#pragma unroll
    for (int ct = 0; ct < 4; ct++) {
      int col = ct * 16 + row;
#pragma unroll
      for (int v = 0; v < 8; v++) stg[(rb + v) * DD + col] = h[ct][v];
    }
    __asm__ __volatile__("" ::: "memory");
    v16bf ha0 = load_a_frag(stg, DD, 0, lane);
    v16bf ha1 = load_a_frag(stg, DD, 32, lane);
#pragma unroll
    for (int ct = 0; ct < 4; ct++) {
      v8f ir, iz, inn, hr, hz, hn;
#pragma unroll
      for (int v = 0; v < 8; v++) {
        ir[v] = bI[ct];  iz[v] = bI[4 + ct];  inn[v] = bI[8 + ct];
        hr[v] = bH[ct];  hz[v] = bH[4 + ct];  hn[v]  = bH[8 + ct];
      }
      ir  = wmma_bf16(xa0, load_b_frag(fih, ct, lane), ir);
      ir  = wmma_bf16(xa1, load_b_frag(fih, 12 + ct, lane), ir);
      iz  = wmma_bf16(xa0, load_b_frag(fih, 4 + ct, lane), iz);
      iz  = wmma_bf16(xa1, load_b_frag(fih, 16 + ct, lane), iz);
      inn = wmma_bf16(xa0, load_b_frag(fih, 8 + ct, lane), inn);
      inn = wmma_bf16(xa1, load_b_frag(fih, 20 + ct, lane), inn);
      hr  = wmma_bf16(ha0, load_b_frag(fhh, ct, lane), hr);
      hr  = wmma_bf16(ha1, load_b_frag(fhh, 12 + ct, lane), hr);
      hz  = wmma_bf16(ha0, load_b_frag(fhh, 4 + ct, lane), hz);
      hz  = wmma_bf16(ha1, load_b_frag(fhh, 16 + ct, lane), hz);
      hn  = wmma_bf16(ha0, load_b_frag(fhh, 8 + ct, lane), hn);
      hn  = wmma_bf16(ha1, load_b_frag(fhh, 20 + ct, lane), hn);
#pragma unroll
      for (int v = 0; v < 8; v++) {
        float r = 1.f / (1.f + __expf(-(ir[v] + hr[v])));
        float z = 1.f / (1.f + __expf(-(iz[v] + hz[v])));
        float n = tanhf(inn[v] + r * hn[v]);
        float hnew = (1.f - z) * n + z * h[ct][v];
        h[ct][v] = (t < sl[v]) ? hnew : h[ct][v];
      }
    }
  }
#pragma unroll
  for (int ct = 0; ct < 4; ct++) {
    int col = ct * 16 + row;
#pragma unroll
    for (int v = 0; v < 8; v++)
      srep[(size_t)(b0 + rb + v) * DD + col] = h[ct][v];
  }
}

// ---------------------------------------------------------------- BPR scoring
__global__ void __launch_bounds__(128) k_score(const float* __restrict__ ufin,
    const float* __restrict__ itfin, const float* __restrict__ srep,
    const int* __restrict__ uidx, const int* __restrict__ pidx,
    const int* __restrict__ nidx, const int* __restrict__ pbeh,
    const float* __restrict__ lng, const float* __restrict__ lnb,
    float* __restrict__ scoreBuf) {
  int gt = blockIdx.x * 128 + threadIdx.x;
  int b = gt >> 5;
  if (b >= BB) return;
  int lane = gt & 31;
  int c = lane * 2;
  size_t uo = (size_t)uidx[b] * DD;
  size_t so = (size_t)b * DD;
  float x0 = ufin[uo + c] + srep[so + c];
  float x1 = ufin[uo + c + 1] + srep[so + c + 1];
  float s = x0 + x1, s2 = x0 * x0 + x1 * x1;
#pragma unroll
  for (int m = 1; m < 32; m <<= 1) { s += __shfl_xor(s, m, 32); s2 += __shfl_xor(s2, m, 32); }
  float mu  = s * (1.f / DD);
  float var = s2 * (1.f / DD) - mu * mu;
  float inv = rsqrtf(var + LN_EPS);
  x0 = (x0 - mu) * inv * lng[c] + lnb[c];
  x1 = (x1 - mu) * inv * lng[c + 1] + lnb[c + 1];
  size_t po = (size_t)pidx[b] * DD, no = (size_t)nidx[b] * DD;
  float p0 = itfin[po + c], p1 = itfin[po + c + 1];
  float n0 = itfin[no + c], n1 = itfin[no + c + 1];
  float dp = x0 * p0 + x1 * p1;
  float dn = x0 * n0 + x1 * n1;
  float nu = x0 * x0 + x1 * x1;
  float np = p0 * p0 + p1 * p1;
  float nn = n0 * n0 + n1 * n1;
#pragma unroll
  for (int m = 1; m < 32; m <<= 1) {
    dp += __shfl_xor(dp, m, 32); dn += __shfl_xor(dn, m, 32);
    nu += __shfl_xor(nu, m, 32); np += __shfl_xor(np, m, 32); nn += __shfl_xor(nn, m, 32);
  }
  if (lane == 0) {
    const float wmap[4] = {1.f, 1.25f, 1.6f, 2.1f};
    int bi = pbeh[b]; bi = bi < 0 ? 0 : (bi > 3 ? 3 : bi);
    float x  = dn - dp;                                     // softplus(neg-pos)
    float sp = (x > 20.f) ? x : log1pf(__expf(x));
    float val = sp * wmap[bi] + REG_CF * (sqrtf(nu) + sqrtf(np) + sqrtf(nn));
    scoreBuf[b] = val * (1.f / BB);
  }
}

__global__ void __launch_bounds__(256) k_reduce(const float* __restrict__ buf, int n,
                                                float* __restrict__ out) {
  __shared__ float sh[256];
  float s = 0.f;
  for (int i = threadIdx.x; i < n; i += 256) s += buf[i];
  sh[threadIdx.x] = s;
  __syncthreads();
  for (int st = 128; st > 0; st >>= 1) {
    if (threadIdx.x < st) sh[threadIdx.x] += sh[threadIdx.x + st];
    __syncthreads();
  }
  if (threadIdx.x == 0) out[0] = sh[0];
}

// ---------------------------------------------------------------- host launch
extern "C" void kernel_launch(void* const* d_in, const int* in_sizes, int n_in,
                              void* d_out, int out_size, void* d_ws, size_t ws_size,
                              hipStream_t stream) {
  (void)in_sizes; (void)n_in; (void)out_size; (void)ws_size;
  const float* user_emb = (const float*)d_in[0];
  const float* item_emb = (const float*)d_in[1];
  const float* beh_emb  = (const float*)d_in[2];
  const float* t_w1     = (const float*)d_in[3];
  const float* t_b1     = (const float*)d_in[4];
  const float* t_w2     = (const float*)d_in[5];
  const float* t_b2     = (const float*)d_in[6];
  const float* gru_w_ih = (const float*)d_in[7];
  const float* gru_w_hh = (const float*)d_in[8];
  const float* gru_b_ih = (const float*)d_in[9];
  const float* gru_b_hh = (const float*)d_in[10];
  const float* gnn_uw   = (const float*)d_in[11];
  const float* gnn_ub   = (const float*)d_in[12];
  const float* gnn_iw   = (const float*)d_in[13];
  const float* gnn_ib   = (const float*)d_in[14];
  const float* ln_g     = (const float*)d_in[15];
  const float* ln_b     = (const float*)d_in[16];
  const float* ui_vals  = (const float*)d_in[17];
  const float* adj_vals = (const float*)d_in[18];
  const float* delta    = (const float*)d_in[19];
  const int* ui_rows    = (const int*)d_in[20];
  const int* ui_cols    = (const int*)d_in[21];
  const int* adj_rows   = (const int*)d_in[22];
  const int* adj_cols   = (const int*)d_in[23];
  const int* seq_items  = (const int*)d_in[24];
  const int* seq_beh    = (const int*)d_in[25];
  const int* seq_len    = (const int*)d_in[26];
  const int* user_idx   = (const int*)d_in[27];
  const int* pos_idx    = (const int*)d_in[28];
  const int* neg_idx    = (const int*)d_in[29];
  const int* pos_beh    = (const int*)d_in[30];

  float* ws   = (float*)d_ws;
  float* u0   = ws;                               // U*D  (layer-0 agg -> layer-0 out)
  float* u1   = u0 + (size_t)UN * DD;             // U*D  (layer-1 agg -> final users)
  float* i0   = u1 + (size_t)UN * DD;             // I*D
  float* i1   = i0 + (size_t)IT_N * DD;           // I*D  (final items)
  float* seqx = i1 + (size_t)IT_N * DD;           // B*T*D
  float* srep = seqx + (size_t)BB * TT * DD;      // B*D
  float* scoreBuf = srep + (size_t)BB * DD;       // B

  const int egUI = (int)(((long)E_UI * 32 + 255) / 256);
  const int egII = (int)(((long)E_II * 32 + 255) / 256);
  const int ubl  = (UN / 16 + 3) / 4;
  const int ibl  = (IT_N / 16 + 3) / 4;

  // ---- GNN layer 0
  hipMemsetAsync(u0, 0, sizeof(float) * (size_t)UN * DD, stream);
  hipMemsetAsync(i0, 0, sizeof(float) * (size_t)IT_N * DD, stream);
  k_edges_ui<<<egUI, 256, 0, stream>>>(user_emb, item_emb, ui_vals, ui_rows, ui_cols,
                                       u0, i0, E_UI);
  k_edges_ii<<<egII, 256, 0, stream>>>(item_emb, adj_vals, adj_rows, adj_cols, i0, E_II);
  k_transform<<<ubl, 128, 0, stream>>>(user_emb, u0, gnn_uw, gnn_ub, UN);
  k_transform<<<ibl, 128, 0, stream>>>(item_emb, i0, gnn_iw, gnn_ib, IT_N);

  // ---- GNN layer 1
  hipMemsetAsync(u1, 0, sizeof(float) * (size_t)UN * DD, stream);
  hipMemsetAsync(i1, 0, sizeof(float) * (size_t)IT_N * DD, stream);
  k_edges_ui<<<egUI, 256, 0, stream>>>(u0, i0, ui_vals, ui_rows, ui_cols, u1, i1, E_UI);
  k_edges_ii<<<egII, 256, 0, stream>>>(i0, adj_vals, adj_rows, adj_cols, i1, E_II);
  k_transform<<<ubl, 128, 0, stream>>>(u0, u1, gnn_uw + DD * DD, gnn_ub + DD, UN);
  k_transform<<<ibl, 128, 0, stream>>>(i0, i1, gnn_iw + DD * DD, gnn_ib + DD, IT_N);

  // ---- sequence features + time MLP + LN
  k_seqx<<<(BB * TT / 16) / 4, 128, 0, stream>>>(i1, beh_emb, seq_items, seq_beh, delta,
                                                 t_w1, t_b1, t_w2, t_b2, ln_g, ln_b, seqx);
  // ---- GRU scan
  k_gru<<<BB / 64, 128, 0, stream>>>(seqx, gru_w_ih, gru_w_hh, gru_b_ih, gru_b_hh,
                                     seq_len, srep);
  // ---- BPR loss
  k_score<<<BB / 4, 128, 0, stream>>>(u1, i1, srep, user_idx, pos_idx, neg_idx, pos_beh,
                                      ln_g, ln_b, scoreBuf);
  k_reduce<<<1, 256, 0, stream>>>(scoreBuf, BB, (float*)d_out);
}
